// biSoftmax_36936718746194
// MI455X (gfx1250) — compile-verified, matrix operand order checked by reading, not run on hardware
//
#include <hip/hip_runtime.h>
#include <math.h>

// ---------------------------------------------------------------------------
// Problem constants (from reference)
// ---------------------------------------------------------------------------
#define HD   8
#define SEQ  2048
#define DM   2048

typedef __attribute__((ext_vector_type(16))) __bf16 v16bf;
typedef __attribute__((ext_vector_type(8)))  float  v8f;

#if defined(__HIP_DEVICE_COMPILE__) && \
    __has_builtin(__builtin_amdgcn_global_load_async_to_lds_b128) && \
    __has_builtin(__builtin_amdgcn_s_wait_asynccnt)
#define HAS_ASYNC_LDS 1
#else
#define HAS_ASYNC_LDS 0
#endif

__device__ __forceinline__ unsigned short f32_to_bf16(float f) {
    unsigned u = __float_as_uint(f);
    u += 0x7FFFu + ((u >> 16) & 1u);          // round-to-nearest-even
    return (unsigned short)(u >> 16);
}

__device__ __forceinline__ void storeOut(float* p, float v)          { *p = v; }
__device__ __forceinline__ void storeOut(unsigned short* p, float v) { *p = f32_to_bf16(v); }

union FragAB {
    v16bf v;
    uint4 q[2];
};

#if HAS_ASYNC_LDS
// 16B global -> LDS async copy (GLOBAL_LOAD_ASYNC_TO_LDS_B128, ASYNCcnt-tracked).
// This clang declares the builtin's pointer params as pointers to a 16B int
// vector in AS(1) (global) and AS(3) (LDS).
typedef int b128vec __attribute__((vector_size(16)));
__device__ __forceinline__ void async_cp16(const unsigned short* g, unsigned short* l) {
    __builtin_amdgcn_global_load_async_to_lds_b128(
        (__attribute__((address_space(1))) b128vec*)(const void*)g,
        (__attribute__((address_space(3))) b128vec*)(void*)l,
        0, 0);
}
#endif

// ---------------------------------------------------------------------------
// Tiled NT GEMM:  C[M,N] = A[M,K] (row major) * B[N,K]^T (row major)
// bf16 inputs, f32 accumulate via v_wmma_f32_16x16x32_bf16.
// Block tile 128x128, BK=64, 256 threads = 8 waves, wave tile 32x64.
// Async path: double-buffered LDS fed by global_load_async_to_lds_b128.
// ---------------------------------------------------------------------------
template <typename OutT>
__global__ __launch_bounds__(256)
void gemm_nt_bf16(const unsigned short* __restrict__ A,
                  const unsigned short* __restrict__ B,
                  OutT* __restrict__ C,
                  int M, int N, int K,
                  long long sAb, long long sBb, long long sCb)
{
    constexpr int BM = 128, BN = 128, BK = 64;
    constexpr int LDT = BK + 8;                 // half-elem stride: 144 B rows (16B aligned)

    const int tid  = threadIdx.x;
    const int lane = tid & 31;
    const int wid  = tid >> 5;
    const int wm   = (wid & 3) * 32;            // wave M offset inside block tile
    const int wn   = (wid >> 2) * 64;           // wave N offset inside block tile
    const int hl   = lane >> 4;                 // half-wave selector (K chunk select)
    const int l16  = lane & 15;                 // row (M or N) within 16x16 tile

    const int m0 = blockIdx.y * BM;
    const int n0 = blockIdx.x * BN;
    const unsigned short* Ah = A + (size_t)blockIdx.z * (size_t)sAb;
    const unsigned short* Bh = B + (size_t)blockIdx.z * (size_t)sBb;
    OutT*                 Ch = C + (size_t)blockIdx.z * (size_t)sCb;

    v8f acc[2][4];
    #pragma unroll
    for (int mi = 0; mi < 2; ++mi)
        #pragma unroll
        for (int ni = 0; ni < 4; ++ni)
            acc[mi][ni] = (v8f){0.f, 0.f, 0.f, 0.f, 0.f, 0.f, 0.f, 0.f};

    // Fragment load per ISA 16-bit A/B layout: lane l16 = row, half selects
    // K chunks [8*hl .. 8*hl+7] and [16+8*hl .. 23+8*hl]  (two ds_load_b128).
    auto fragLd = [&](const unsigned short* L, int rowBase, int k) -> FragAB {
        FragAB f;
        const unsigned short* p = &L[(rowBase + l16) * LDT + k + 8 * hl];
        f.q[0] = *reinterpret_cast<const uint4*>(p);
        f.q[1] = *reinterpret_cast<const uint4*>(p + 16);
        return f;
    };
    auto computeTile = [&](const unsigned short* TA, const unsigned short* TB) {
        #pragma unroll
        for (int ks = 0; ks < BK; ks += 32) {
            FragAB af[2], bf[4];
            #pragma unroll
            for (int mi = 0; mi < 2; ++mi) af[mi] = fragLd(TA, wm + mi * 16, ks);
            #pragma unroll
            for (int ni = 0; ni < 4; ++ni) bf[ni] = fragLd(TB, wn + ni * 16, ks);
            #pragma unroll
            for (int mi = 0; mi < 2; ++mi)
                #pragma unroll
                for (int ni = 0; ni < 4; ++ni)
                    acc[mi][ni] = __builtin_amdgcn_wmma_f32_16x16x32_bf16(
                        false, af[mi].v, false, bf[ni].v,
                        (short)0, acc[mi][ni], false, false);
        }
    };

    const int nkt = K / BK;

#if HAS_ASYNC_LDS
    // ---- async double-buffered path: no VGPR staging, copies overlap math ----
    __shared__ unsigned short lA[2][BM * LDT];
    __shared__ unsigned short lB[2][BN * LDT];

    // Each wave issues exactly 8 async b128 ops per K-tile (4 for A, 4 for B),
    // so after issuing the next tile, s_wait_asynccnt(8) == "previous tile done"
    // (async loads complete in order).
    auto issueTile = [&](int kt, int buf) {
        #pragma unroll
        for (int j = 0; j < 4; ++j) {
            const int c = wid * 128 + j * 32 + lane;   // chunk id 0..1023
            const int row = c >> 3, col = (c & 7) * 8;
            async_cp16(Ah + (size_t)(m0 + row) * (size_t)K + (size_t)kt * BK + col,
                       &lA[buf][row * LDT + col]);
        }
        #pragma unroll
        for (int j = 0; j < 4; ++j) {
            const int c = wid * 128 + j * 32 + lane;
            const int row = c >> 3, col = (c & 7) * 8;
            async_cp16(Bh + (size_t)(n0 + row) * (size_t)K + (size_t)kt * BK + col,
                       &lB[buf][row * LDT + col]);
        }
    };

    issueTile(0, 0);
    for (int kt = 0; kt < nkt; ++kt) {
        __syncthreads();                       // all waves done reading buf^1
        if (kt + 1 < nkt) {
            issueTile(kt + 1, (kt + 1) & 1);   // overlap next tile with math
            __builtin_amdgcn_s_wait_asynccnt(8);
        } else {
            __builtin_amdgcn_s_wait_asynccnt(0);
        }
        __syncthreads();                       // tile kt landed for every wave
        computeTile(&lA[kt & 1][0], &lB[kt & 1][0]);
    }
#else
    // ---- fallback: register-staged single-buffer path ----
    __shared__ unsigned short lA[BM * LDT];
    __shared__ unsigned short lB[BN * LDT];

    const int crow = tid >> 3;                  // 0..31
    const int ccol = (tid & 7) * 8;             // half-element column
    uint4 ra[4], rb[4];

    auto gload = [&](const unsigned short* __restrict__ P, int rowBase, int kt, uint4* r) {
        #pragma unroll
        for (int j = 0; j < 4; ++j) {
            const unsigned short* p = P + (size_t)(rowBase + crow + j * 32) * (size_t)K
                                        + (size_t)kt * BK + ccol;
            r[j] = *reinterpret_cast<const uint4*>(p);
        }
    };
    auto sstore = [&](unsigned short* L, const uint4* r) {
        #pragma unroll
        for (int j = 0; j < 4; ++j)
            *reinterpret_cast<uint4*>(&L[(crow + j * 32) * LDT + ccol]) = r[j];
    };

    gload(Ah, m0, 0, ra);
    gload(Bh, n0, 0, rb);
    for (int kt = 0; kt < nkt; ++kt) {
        __syncthreads();
        sstore(lA, ra);
        sstore(lB, rb);
        __syncthreads();
        if (kt + 1 < nkt) {
            gload(Ah, m0, kt + 1, ra);
            gload(Bh, n0, kt + 1, rb);
        }
        computeTile(lA, lB);
    }
#endif

    // Epilogue per C layout: VGPR r, lane -> row = r + 8*hl, col = l16
    #pragma unroll
    for (int mi = 0; mi < 2; ++mi)
        #pragma unroll
        for (int ni = 0; ni < 4; ++ni) {
            const int col = n0 + wn + ni * 16 + l16;
            #pragma unroll
            for (int r = 0; r < 8; ++r) {
                const int row = m0 + wm + mi * 16 + r + 8 * hl;
                storeOut(&Ch[(size_t)row * (size_t)N + col], acc[mi][ni][r]);
            }
        }
}

// ---------------------------------------------------------------------------
// f32 -> bf16 convert (vectorized x4)
// ---------------------------------------------------------------------------
__global__ __launch_bounds__(256)
void cvt_f32_bf16_kernel(const float* __restrict__ in, unsigned short* __restrict__ out,
                         size_t n4)
{
    size_t i = (size_t)blockIdx.x * blockDim.x + threadIdx.x;
    if (i >= n4) return;
    float4 f = reinterpret_cast<const float4*>(in)[i];
    union { unsigned short u[4]; uint2 v; } r;
    r.u[0] = f32_to_bf16(f.x); r.u[1] = f32_to_bf16(f.y);
    r.u[2] = f32_to_bf16(f.z); r.u[3] = f32_to_bf16(f.w);
    reinterpret_cast<uint2*>(out)[i] = r.v;
}

// ---------------------------------------------------------------------------
// circle = (s1 - s2)^2, f32 in -> bf16 out (vectorized x4)
// ---------------------------------------------------------------------------
__global__ __launch_bounds__(256)
void sqdiff_kernel(const float* __restrict__ a, const float* __restrict__ b,
                   unsigned short* __restrict__ out, size_t n4)
{
    size_t i = (size_t)blockIdx.x * blockDim.x + threadIdx.x;
    if (i >= n4) return;
    float4 x = reinterpret_cast<const float4*>(a)[i];
    float4 y = reinterpret_cast<const float4*>(b)[i];
    float dx = x.x - y.x, dy = x.y - y.y, dz = x.z - y.z, dw = x.w - y.w;
    union { unsigned short u[4]; uint2 v; } r;
    r.u[0] = f32_to_bf16(dx * dx); r.u[1] = f32_to_bf16(dy * dy);
    r.u[2] = f32_to_bf16(dz * dz); r.u[3] = f32_to_bf16(dw * dw);
    reinterpret_cast<uint2*>(out)[i] = r.v;
}

// ---------------------------------------------------------------------------
// Row softmax with scale: out = bf16(softmax(scale * row)), row length SEQ
// one 256-thread block per row, 8 elems/thread
// ---------------------------------------------------------------------------
__global__ __launch_bounds__(256)
void softmax_row_kernel(const float* __restrict__ in, unsigned short* __restrict__ out,
                        float scale)
{
    const size_t row = blockIdx.x;
    const float* p = in + row * (size_t)SEQ;
    unsigned short* o = out + row * (size_t)SEQ;
    const int tid = threadIdx.x;

    __shared__ float red[8];
    __shared__ float sM, sS;

    float v[8];
    float m = -3.4e38f;
    #pragma unroll
    for (int i = 0; i < 8; ++i) { v[i] = p[tid + i * 256]; m = fmaxf(m, v[i]); }

    #pragma unroll
    for (int off = 16; off; off >>= 1) m = fmaxf(m, __shfl_xor(m, off, 32));
    if ((tid & 31) == 0) red[tid >> 5] = m;
    __syncthreads();
    if (tid == 0) {
        float t = red[0];
        #pragma unroll
        for (int i = 1; i < 8; ++i) t = fmaxf(t, red[i]);
        sM = t;
    }
    __syncthreads();
    m = sM;

    float s = 0.f;
    #pragma unroll
    for (int i = 0; i < 8; ++i) { v[i] = __expf(scale * (v[i] - m)); s += v[i]; }

    #pragma unroll
    for (int off = 16; off; off >>= 1) s += __shfl_xor(s, off, 32);
    if ((tid & 31) == 0) red[tid >> 5] = s;
    __syncthreads();
    if (tid == 0) {
        float t = 0.f;
        #pragma unroll
        for (int i = 0; i < 8; ++i) t += red[i];
        sS = t;
    }
    __syncthreads();
    const float inv = 1.0f / sS;

    #pragma unroll
    for (int i = 0; i < 8; ++i) o[tid + i * 256] = f32_to_bf16(v[i] * inv);
}

// ---------------------------------------------------------------------------
// Batched bf16 transpose (SEQ x DM per head), 32x32 LDS tiles
// ---------------------------------------------------------------------------
__global__ __launch_bounds__(256)
void transpose_bf16_kernel(const unsigned short* __restrict__ in,
                           unsigned short* __restrict__ out)
{
    __shared__ unsigned short t[32][33];
    const size_t bs = (size_t)SEQ * DM;
    const unsigned short* I = in + (size_t)blockIdx.z * bs;
    unsigned short*       O = out + (size_t)blockIdx.z * bs;
    const int tx = threadIdx.x & 31, ty = threadIdx.x >> 5;
    const int bx = blockIdx.x * 32, by = blockIdx.y * 32;
    #pragma unroll
    for (int i = 0; i < 32; i += 8)
        t[ty + i][tx] = I[(size_t)(by + ty + i) * DM + bx + tx];
    __syncthreads();
    #pragma unroll
    for (int i = 0; i < 32; i += 8)
        O[(size_t)(bx + ty + i) * SEQ + by + tx] = t[tx][ty + i];
}

// ---------------------------------------------------------------------------
// Host-side orchestration
// ---------------------------------------------------------------------------
extern "C" void kernel_launch(void* const* d_in, const int* in_sizes, int n_in,
                              void* d_out, int out_size, void* d_ws, size_t ws_size,
                              hipStream_t stream)
{
    (void)in_sizes; (void)n_in; (void)out_size; (void)ws_size;

    const float* x = (const float*)d_in[0];
    const float* W[7];
    for (int i = 0; i < 7; ++i) W[i] = (const float*)d_in[1 + i];
    float* out = (float*)d_out;

    const size_t SD  = (size_t)SEQ * DM;    // 4,194,304 (also SEQ*SEQ here)
    const size_t HSD = (size_t)HD * SD;     // 33,554,432

    char* ws = (char*)d_ws;
    size_t off = 0;
    auto alloc = [&](size_t bytes) -> void* {
        void* p = ws + off;
        off += (bytes + 255) & ~(size_t)255;
        return p;
    };

    unsigned short* xb  = (unsigned short*)alloc(HSD * 2);
    unsigned short* wb[7];
    for (int i = 0; i < 7; ++i) wb[i] = (unsigned short*)alloc(SD * 2);
    unsigned short* q1b = (unsigned short*)alloc(HSD * 2);
    unsigned short* k1b = (unsigned short*)alloc(HSD * 2);
    unsigned short* q2b = (unsigned short*)alloc(HSD * 2);
    unsigned short* k2b = (unsigned short*)alloc(HSD * 2);
    unsigned short* vb  = (unsigned short*)alloc(HSD * 2);
    unsigned short* vtb = (unsigned short*)alloc(HSD * 2);
    float* lg1 = (float*)alloc(HSD * 4);    // logits1, later score1
    float* lg2 = (float*)alloc(HSD * 4);    // logits2, later score2
    unsigned short* a1b = (unsigned short*)alloc(HSD * 2);
    unsigned short* a2b = (unsigned short*)alloc(HSD * 2);
    unsigned short* circ = a1b;             // reuse: a1 dead after score1 GEMM

    const dim3 blk(256);

    // 1) convert inputs to bf16
    cvt_f32_bf16_kernel<<<dim3((unsigned)(HSD / 4 / 256)), blk, 0, stream>>>(x, xb, HSD / 4);
    for (int i = 0; i < 7; ++i)
        cvt_f32_bf16_kernel<<<dim3((unsigned)(SD / 4 / 256)), blk, 0, stream>>>(W[i], wb[i], SD / 4);

    // 2) projections: [16384,2048] @ W^T  (heads fused in M; weights shared)
    const dim3 gProj(DM / 128, (HD * SEQ) / 128, 1);
    gemm_nt_bf16<unsigned short><<<gProj, blk, 0, stream>>>(xb, wb[0], q1b, HD * SEQ, DM, DM, 0, 0, 0);
    gemm_nt_bf16<unsigned short><<<gProj, blk, 0, stream>>>(xb, wb[1], k1b, HD * SEQ, DM, DM, 0, 0, 0);
    gemm_nt_bf16<unsigned short><<<gProj, blk, 0, stream>>>(xb, wb[2], q2b, HD * SEQ, DM, DM, 0, 0, 0);
    gemm_nt_bf16<unsigned short><<<gProj, blk, 0, stream>>>(xb, wb[3], k2b, HD * SEQ, DM, DM, 0, 0, 0);
    gemm_nt_bf16<unsigned short><<<gProj, blk, 0, stream>>>(xb, wb[4], vb,  HD * SEQ, DM, DM, 0, 0, 0);

    // 3) attention logits per head: q @ k^T
    const dim3 gAtt(SEQ / 128, SEQ / 128, HD);
    gemm_nt_bf16<float><<<gAtt, blk, 0, stream>>>(q1b, k1b, lg1, SEQ, SEQ, DM,
                                                  (long long)SD, (long long)SD, (long long)SD);
    gemm_nt_bf16<float><<<gAtt, blk, 0, stream>>>(q2b, k2b, lg2, SEQ, SEQ, DM,
                                                  (long long)SD, (long long)SD, (long long)SD);

    // 4) softmax(scale * logits) -> bf16
    const float scale = 1.0f / sqrtf((float)SEQ);
    softmax_row_kernel<<<dim3(HD * SEQ), blk, 0, stream>>>(lg1, a1b, scale);
    softmax_row_kernel<<<dim3(HD * SEQ), blk, 0, stream>>>(lg2, a2b, scale);

    // 5) scores: a @ W^T (W1/W2 shared over heads) -> f32 (reuse logits buffers)
    gemm_nt_bf16<float><<<gAtt, blk, 0, stream>>>(a1b, wb[5], lg1, SEQ, SEQ, SEQ,
                                                  (long long)SD, 0LL, (long long)SD);
    gemm_nt_bf16<float><<<gAtt, blk, 0, stream>>>(a2b, wb[6], lg2, SEQ, SEQ, SEQ,
                                                  (long long)SD, 0LL, (long long)SD);

    // 6) circle = (score1 - score2)^2 -> bf16 (into a1b storage)
    sqdiff_kernel<<<dim3((unsigned)(HSD / 4 / 256)), blk, 0, stream>>>(lg1, lg2, circ, HSD / 4);

    // 7) transpose v per head so the final GEMM is NT
    transpose_bf16_kernel<<<dim3(DM / 32, SEQ / 32, HD), blk, 0, stream>>>(vb, vtb);

    // 8) out = circle @ v  (== circle @ (v^T)^T, NT with vtb)
    gemm_nt_bf16<float><<<gAtt, blk, 0, stream>>>(circ, vtb, out, SEQ, DM, SEQ,
                                                  (long long)SD, (long long)SD, (long long)SD);
}
